// OptimizerNetwork_76364518523026
// MI455X (gfx1250) — matrix operand order repaired and checked by reading, not compiled, not run on hardware
//
#include <hip/hip_runtime.h>

typedef _Float16 f16;
typedef __attribute__((ext_vector_type(16))) _Float16 v16h;
typedef __attribute__((ext_vector_type(8)))  float    v8f;

#define HID 20
#define GAT 80   // 4*HID

#if __has_builtin(__builtin_amdgcn_tanhf)
__device__ __forceinline__ float fast_tanh(float x) { return __builtin_amdgcn_tanhf(x); }
#else
__device__ __forceinline__ float fast_tanh(float x) {
    // tanh(x) = 1 - 2/(exp(2x)+1)
    float e = __expf(2.f * x);
    return 1.f - 2.f / (e + 1.f);
}
#endif
__device__ __forceinline__ float fast_sigmoid(float x) {
    return 0.5f + 0.5f * fast_tanh(0.5f * x);
}

__launch_bounds__(128)
__global__ void lstm_opt_kernel(
    const float* __restrict__ inp,
    const float* __restrict__ h0,  const float* __restrict__ h1,
    const float* __restrict__ c0,  const float* __restrict__ c1,
    const float* __restrict__ Wih1, const float* __restrict__ Whh1,
    const float* __restrict__ bih1, const float* __restrict__ bhh1,
    const float* __restrict__ Wih2, const float* __restrict__ Whh2,
    const float* __restrict__ bih2, const float* __restrict__ bhh2,
    const float* __restrict__ Wout, const float* __restrict__ bout,
    float* __restrict__ out, int rows)
{
    // ---- LDS ----
    __shared__ __align__(32) f16 sB[15][32][16];     // B fragments, lane-major
    __shared__ float sBias1[GAT], sBias2[GAT], sWout[HID];
    __shared__ float sBout;
    __shared__ float sXa[4][16], sXb[4][16];
    __shared__ __align__(16) float sH0[4][16*HID], sH1[4][16*HID];
    __shared__ float sHn0[4][16*HID], sHn1[4][16*HID];
    __shared__ float sG[4][16*GAT];

    const int tid = threadIdx.x;

    // ---- Stage shared weights as WMMA B-fragments (once per block) ----
    // B 32x16 f16 layout: lane L, f16-slot s  <->  B[K = (s<8 ? s : s+8) + 8*(L>=16)][N = L%16]
    for (int idx = tid; idx < 15*32*16; idx += 128) {
        int f   = idx >> 9;          // fragment 0..14
        int rem = idx & 511;
        int L   = rem >> 4;          // lane 0..31
        int s   = rem & 15;          // slot 0..15
        int K   = ((s < 8) ? s : s + 8) + 8*(L >> 4);   // 0..31
        int n   = L & 15;            // column within tile
        int t   = f % 5, grp = f / 5;
        int g   = 16*t + n;          // gate column 0..79
        float val = 0.f;
        if (grp == 0) {              // cell1: rows = [Wih1^T(2) ; Whh1^T(20) ; 0]
            if      (K == 0)       val = Wih1[g*2 + 0];
            else if (K == 1)       val = Wih1[g*2 + 1];
            else if (K < 2 + HID)  val = Whh1[g*HID + (K - 2)];
        } else if (grp == 1) {       // cell2 input path: Wih2^T
            if (K < HID)           val = Wih2[g*HID + K];
        } else {                     // cell2 hidden path: Whh2^T
            if (K < HID)           val = Whh2[g*HID + K];
        }
        sB[f][L][s] = (f16)val;
    }
    if (tid < GAT) { sBias1[tid] = bih1[tid] + bhh1[tid];
                     sBias2[tid] = bih2[tid] + bhh2[tid]; }
    if (tid < HID) sWout[tid] = Wout[tid];
    if (tid == 0)  sBout = bout[0];

    // ---- Per-wave tile of 16 rows ----
    const int  wv   = tid >> 5;
    const int  lane = tid & 31;
    const long r0   = ((long)blockIdx.x * 4 + wv) * 16;
    const int  Mcol = lane & 15;          // N column in D; also row id for lane pair
    const int  half = lane >> 4;          // 0 or 1
    const int  mofs = 8 * half;           // D row offset for this lane half

    // preprocess gradients (lanes 0..15 handle the 16 rows)
    if (lane < 16) {
        long r = r0 + lane; if (r >= rows) r = rows - 1;
        float x  = inp[r];
        float ax = fabsf(x);
        const float thr = 4.5399929762484854e-05f;   // exp(-10)
        float a, b;
        if (ax >= thr) {
            a = logf(ax + 1e-8f) * 0.1f;
            b = (x > 0.f) ? 1.f : ((x < 0.f) ? -1.f : 0.f);
        } else {
            a = -1.f;
            b = 22026.465794806718f * x;             // exp(10)*x
        }
        sXa[wv][lane] = a; sXb[wv][lane] = b;
    }
    // stage h0/h1 tiles: a 16-row tile is a contiguous 320-float run
    if (r0 + 16 <= rows) {
        const float4* h0v = (const float4*)(h0 + r0 * HID);
        const float4* h1v = (const float4*)(h1 + r0 * HID);
        float4* d0 = (float4*)&sH0[wv][0];
        float4* d1 = (float4*)&sH1[wv][0];
        for (int i = lane; i < (16*HID)/4; i += 32) {   // 80 float4s
            d0[i] = h0v[i];
            d1[i] = h1v[i];
        }
    } else {
        for (int i = lane; i < 16*HID; i += 32) {
            long rr = r0 + i / HID; if (rr >= rows) rr = rows - 1;
            long off = rr * HID + (i % HID);
            sH0[wv][i] = h0[off];
            sH1[wv][i] = h1[off];
        }
    }
    __syncthreads();

    // ---- Build A1 (16x32 f16): cols 0=a, 1=b, 2..21=h0, rest 0 ----
    // A layout: lane half 0 -> K {0..7,16..23}; half 1 -> K {8..15,24..31}
    v16h a1;
    #pragma unroll
    for (int s = 0; s < 16; ++s) {
        int K = ((s < 8) ? s : s + 8) + 8 * half;
        float v = 0.f;
        if      (K == 0)       v = sXa[wv][Mcol];
        else if (K == 1)       v = sXb[wv][Mcol];
        else if (K < 2 + HID)  v = sH0[wv][Mcol*HID + (K - 2)];
        a1[s] = (f16)v;
    }
    // ---- Cell-1 gates: 5 WMMAs over the 80 gate columns ----
    #pragma unroll
    for (int t = 0; t < 5; ++t) {
        v16h b = *(const v16h*)(&sB[t][lane][0]);
        float bias = sBias1[16*t + Mcol];
        v8f acc = {bias,bias,bias,bias,bias,bias,bias,bias};
        acc = __builtin_amdgcn_wmma_f32_16x16x32_f16(false, a1, false, b,
                                                     (short)0, acc, false, false);
        #pragma unroll
        for (int v = 0; v < 8; ++v)
            sG[wv][(v + mofs)*GAT + 16*t + Mcol] = acc[v];
    }
    __syncthreads();

    // ---- Element-wise LSTM cell 1 ----
    float* outH0 = out + (long)rows;
    float* outH1 = out + (long)rows * 21;
    float* outC0 = out + (long)rows * 41;
    float* outC1 = out + (long)rows * 61;
    #pragma unroll
    for (int it = 0; it < 10; ++it) {
        int idx = it*32 + lane;              // 0..319 == M*HID + j
        int M = idx / HID, j = idx % HID;
        long r  = r0 + M;
        long rc = (r >= rows) ? (rows - 1) : r;
        float gi = sG[wv][M*GAT +      j];
        float gf = sG[wv][M*GAT + 20 + j];
        float gg = sG[wv][M*GAT + 40 + j];
        float go = sG[wv][M*GAT + 60 + j];
        float ii = fast_sigmoid(gi);
        float ff = fast_sigmoid(gf);
        float g2 = fast_tanh(gg);
        float oo = fast_sigmoid(go);
        float cn = ff * c0[rc*HID + j] + ii * g2;
        float hn = oo * fast_tanh(cn);
        sHn0[wv][idx] = hn;
        if (r < rows) { outH0[r*HID + j] = hn; outC0[r*HID + j] = cn; }
    }
    __syncthreads();

    // ---- Build A2a (h0n) and A2b (h1), cols 0..19, rest 0 ----
    v16h a2a, a2b;
    #pragma unroll
    for (int s = 0; s < 16; ++s) {
        int K = ((s < 8) ? s : s + 8) + 8 * half;
        float va = 0.f, vb = 0.f;
        if (K < HID) { va = sHn0[wv][Mcol*HID + K]; vb = sH1[wv][Mcol*HID + K]; }
        a2a[s] = (f16)va; a2b[s] = (f16)vb;
    }
    // ---- Cell-2 gates: 2 accumulating WMMAs per tile, 5 tiles ----
    #pragma unroll
    for (int t = 0; t < 5; ++t) {
        v16h bA = *(const v16h*)(&sB[5  + t][lane][0]);
        v16h bB = *(const v16h*)(&sB[10 + t][lane][0]);
        float bias = sBias2[16*t + Mcol];
        v8f acc = {bias,bias,bias,bias,bias,bias,bias,bias};
        acc = __builtin_amdgcn_wmma_f32_16x16x32_f16(false, a2a, false, bA,
                                                     (short)0, acc, false, false);
        acc = __builtin_amdgcn_wmma_f32_16x16x32_f16(false, a2b, false, bB,
                                                     (short)0, acc, false, false);
        #pragma unroll
        for (int v = 0; v < 8; ++v)
            sG[wv][(v + mofs)*GAT + 16*t + Mcol] = acc[v];
    }
    __syncthreads();

    // ---- Element-wise LSTM cell 2 ----
    #pragma unroll
    for (int it = 0; it < 10; ++it) {
        int idx = it*32 + lane;
        int M = idx / HID, j = idx % HID;
        long r  = r0 + M;
        long rc = (r >= rows) ? (rows - 1) : r;
        float gi = sG[wv][M*GAT +      j];
        float gf = sG[wv][M*GAT + 20 + j];
        float gg = sG[wv][M*GAT + 40 + j];
        float go = sG[wv][M*GAT + 60 + j];
        float ii = fast_sigmoid(gi);
        float ff = fast_sigmoid(gf);
        float g2 = fast_tanh(gg);
        float oo = fast_sigmoid(go);
        float cn = ff * c1[rc*HID + j] + ii * g2;
        float hn = oo * fast_tanh(cn);
        sHn1[wv][idx] = hn;
        if (r < rows) { outH1[r*HID + j] = hn; outC1[r*HID + j] = cn; }
    }
    __syncthreads();

    // ---- Output linear: out = h1n . Wout + bout ----
    if (lane < 16) {
        long r = r0 + lane;
        if (r < rows) {
            float s = sBout;
            #pragma unroll
            for (int j = 0; j < HID; ++j)
                s += sHn1[wv][lane*HID + j] * sWout[j];
            out[r] = s;
        }
    }
}

extern "C" void kernel_launch(void* const* d_in, const int* in_sizes, int n_in,
                              void* d_out, int out_size, void* d_ws, size_t ws_size,
                              hipStream_t stream) {
    const float* inp  = (const float*)d_in[0];
    const float* h0   = (const float*)d_in[1];
    const float* h1   = (const float*)d_in[2];
    const float* c0   = (const float*)d_in[3];
    const float* c1   = (const float*)d_in[4];
    const float* Wih1 = (const float*)d_in[5];
    const float* Whh1 = (const float*)d_in[6];
    const float* bih1 = (const float*)d_in[7];
    const float* bhh1 = (const float*)d_in[8];
    const float* Wih2 = (const float*)d_in[9];
    const float* Whh2 = (const float*)d_in[10];
    const float* bih2 = (const float*)d_in[11];
    const float* bhh2 = (const float*)d_in[12];
    const float* Wout = (const float*)d_in[13];
    const float* bout = (const float*)d_in[14];
    float* out = (float*)d_out;
    int rows = in_sizes[0];                 // N (inp is [N,1])

    int tiles  = (rows + 15) / 16;          // 16 rows per wave
    int blocks = (tiles + 3) / 4;           // 4 waves per block
    lstm_opt_kernel<<<blocks, 128, 0, stream>>>(
        inp, h0, h1, c0, c1,
        Wih1, Whh1, bih1, bhh1,
        Wih2, Whh2, bih2, bhh2,
        Wout, bout, out, rows);
}
